// State_Predict_81140522156856
// MI455X (gfx1250) — matrix operand order repaired and checked by reading, not compiled, not run on hardware
//
#include <hip/hip_runtime.h>

// ---------------------------------------------------------------------------
// Problem constants (match the reference)
// ---------------------------------------------------------------------------
#define BATCH 4096
#define SDIM  512
#define ADIM  128
#define DELAYN 16
#define CIN   (2 * SDIM + ADIM)   // 1152
#define G3    (3 * SDIM)          // 1536

typedef unsigned short u16;
typedef __attribute__((ext_vector_type(16))) __bf16        v16bf;
typedef __attribute__((ext_vector_type(8)))  float         v8f;
typedef __attribute__((ext_vector_type(4)))  unsigned int  v4u;

union FragU {
    v16bf v;
    v4u   u[2];
};

// round-to-nearest-even f32 -> bf16 (stored as u16)
__device__ __forceinline__ u16 f2bf(float f) {
    union { float f; unsigned u; } c; c.f = f;
    unsigned r = c.u + 0x7FFFu + ((c.u >> 16) & 1u);
    return (u16)(r >> 16);
}

__device__ __forceinline__ v8f wmma_bf16(v16bf a, v16bf b, v8f c) {
    // (neg_a, A, neg_b, B, c_mod, C, reuse_a, reuse_b)
    return __builtin_amdgcn_wmma_f32_16x16x32_bf16(false, a, false, b,
                                                   (short)0, c, false, false);
}

// A fragment: 16x32 (MxK), row-major memory [rows, stride] of bf16.
// lanes 0-15: M=M0+lane, elems 0-7 = K k0..k0+7, elems 8-15 = K k0+16..k0+23
// lanes 16-31: same M, K offset +8 (per CDNA5 ISA 7.12.2 16-bit A layout)
__device__ __forceinline__ v16bf load_a(const u16* __restrict__ base, int stride,
                                        int M0, int k0, int lane) {
    const int m    = M0 + (lane & 15);
    const int half = lane >> 4;
    const u16* p = base + (size_t)m * stride + k0 + half * 8;
    FragU f;
    f.u[0] = *(const v4u*)(p);       // 8 bf16 = 16B
    f.u[1] = *(const v4u*)(p + 16);  // K+16 block
    return f.v;
}

// B fragment: 32x16 (KxN) where B[k][n] = W[N0+n][k0+k], W row-major [N, stride].
// lanes 0-15: N=N0+lane, elems 0-7 = K k0..k0+7, elems 8-15 = K k0+16..k0+23; lanes 16-31: K offset +8
__device__ __forceinline__ v16bf load_b(const u16* __restrict__ W, int stride,
                                        int N0, int k0, int lane) {
    const int n = N0 + (lane & 15);
    const u16* p = W + (size_t)n * stride + k0 + (lane >> 4) * 8;
    FragU f;
    f.u[0] = *(const v4u*)(p);
    f.u[1] = *(const v4u*)(p + 16);
    return f.v;
}

// ---------------------------------------------------------------------------
// Prep kernels
// ---------------------------------------------------------------------------
__global__ void f32_to_bf16_kernel(const float* __restrict__ src,
                                   u16* __restrict__ dst, int n) {
    int i = blockIdx.x * blockDim.x + threadIdx.x;
    if (i < n) dst[i] = f2bf(src[i]);
}

__global__ void copy_f32_kernel(const float* __restrict__ src,
                                float* __restrict__ dst, int n) {
    int i = blockIdx.x * blockDim.x + threadIdx.x;
    if (i < n) dst[i] = src[i];
}

// h_f32[i,j] = hidden0[j]; h_bf likewise (broadcast initial hidden)
__global__ void init_h_kernel(const float* __restrict__ hidden0,
                              float* __restrict__ h_f32,
                              u16* __restrict__ h_bf, int n) {
    int i = blockIdx.x * blockDim.x + threadIdx.x;
    if (i < n) {
        float v = hidden0[i & (SDIM - 1)];
        h_f32[i] = v;
        h_bf[i]  = f2bf(v);
    }
}

// ---------------------------------------------------------------------------
// GRU step: h_new = (1-z)*tanh(i_n + r*h_n) + z*h_old
//   gates from x@W_ih^T + h@W_hh^T (+biases). r,z accumulate both GEMMs
//   into one accumulator; i_n / h_n kept separate.
// Block: 256 thr = 8 waves; block tile M=64 x J=64.
// Each wave computes TWO 16x16 tiles (M0, M0+16) x J0 so every weight (B)
// fragment feeds two WMMAs -> 20 b128 loads per 12 WMMAs in steady state.
// ---------------------------------------------------------------------------
__global__ __launch_bounds__(256) void gru_kernel(
    const u16* __restrict__ x_bf,    // [B, S] GEMM input x
    const u16* __restrict__ h_bf,    // [B, S] GEMM input h (prev hidden)
    const u16* __restrict__ Wih,     // [3S, S] bf16
    const u16* __restrict__ Whh,     // [3S, S] bf16
    const float* __restrict__ b_ih,  // [3S]
    const float* __restrict__ b_hh,  // [3S]
    float* __restrict__ h_f32,       // [B, S] in: h_old, out: h_new (in-place)
    u16* __restrict__ h_out_bf)      // [B, S] bf16 copy of h_new
{
    const int lane = threadIdx.x & 31;
    const int wid  = threadIdx.x >> 5;
    const int M0 = blockIdx.x * 64 + (wid >> 2) * 32;   // wave: rows M0..M0+31
    const int J0 = blockIdx.y * 64 + (wid & 3) * 16;

    v8f r0 = {}, r1 = {}, z0 = {}, z1 = {};
    v8f in0 = {}, in1 = {}, hn0 = {}, hn1 = {};
    for (int k0 = 0; k0 < SDIM; k0 += 32) {
        v16bf ax0 = load_a(x_bf, SDIM, M0,      k0, lane);
        v16bf ax1 = load_a(x_bf, SDIM, M0 + 16, k0, lane);
        v16bf ah0 = load_a(h_bf, SDIM, M0,      k0, lane);
        v16bf ah1 = load_a(h_bf, SDIM, M0 + 16, k0, lane);
        v16bf bri = load_b(Wih, SDIM, J0,            k0, lane);
        v16bf bzi = load_b(Wih, SDIM, J0 + SDIM,     k0, lane);
        v16bf bni = load_b(Wih, SDIM, J0 + 2 * SDIM, k0, lane);
        v16bf brh = load_b(Whh, SDIM, J0,            k0, lane);
        v16bf bzh = load_b(Whh, SDIM, J0 + SDIM,     k0, lane);
        v16bf bnh = load_b(Whh, SDIM, J0 + 2 * SDIM, k0, lane);
        r0  = wmma_bf16(ax0, bri, r0);   r1  = wmma_bf16(ax1, bri, r1);
        r0  = wmma_bf16(ah0, brh, r0);   r1  = wmma_bf16(ah1, brh, r1);
        z0  = wmma_bf16(ax0, bzi, z0);   z1  = wmma_bf16(ax1, bzi, z1);
        z0  = wmma_bf16(ah0, bzh, z0);   z1  = wmma_bf16(ah1, bzh, z1);
        in0 = wmma_bf16(ax0, bni, in0);  in1 = wmma_bf16(ax1, bni, in1);
        hn0 = wmma_bf16(ah0, bnh, hn0);  hn1 = wmma_bf16(ah1, bnh, hn1);
    }

    // C/D layout: VGPR v -> M = v + 8*(lane>=16), N = lane&15
    const int nn   = lane & 15;
    const float br   = b_ih[J0 + nn] + b_hh[J0 + nn];
    const float bz   = b_ih[SDIM + J0 + nn] + b_hh[SDIM + J0 + nn];
    const float bn_i = b_ih[2 * SDIM + J0 + nn];
    const float bn_h = b_hh[2 * SDIM + J0 + nn];
#pragma unroll
    for (int tile = 0; tile < 2; ++tile) {
        const v8f& ar  = tile ? r1  : r0;
        const v8f& az  = tile ? z1  : z0;
        const v8f& ain = tile ? in1 : in0;
        const v8f& ahn = tile ? hn1 : hn0;
        const int mbase = M0 + tile * 16 + (lane >> 4) * 8;
#pragma unroll
        for (int v = 0; v < 8; ++v) {
            size_t idx = (size_t)(mbase + v) * SDIM + J0 + nn;
            float r  = 1.f / (1.f + __expf(-(ar[v] + br)));
            float z  = 1.f / (1.f + __expf(-(az[v] + bz)));
            float ng = tanhf(ain[v] + bn_i + r * (ahn[v] + bn_h));
            float ho = h_f32[idx];               // same-element read-then-write
            float hn = (1.f - z) * ng + z * ho;
            h_f32[idx]    = hn;
            h_out_bf[idx] = f2bf(hn);
        }
    }
}

// ---------------------------------------------------------------------------
// DNF step: g = [st | a_t | h] @ Wdnf^T + b; st_new = F*(st+D) + (1-F)*N
// K = 1152 handled as virtual concat (each 32-chunk lies in one segment).
// Same 2-M-tile register blocking: 10 b128 loads per 6 WMMAs.
// ---------------------------------------------------------------------------
__global__ __launch_bounds__(256) void dnf_kernel(
    const u16* __restrict__ st_bf,   // [B, S]
    const u16* __restrict__ act_bf,  // [B, DELAY, A]
    int t,
    const u16* __restrict__ hh_bf,   // [B, S]
    const u16* __restrict__ Wdnf,    // [3S, CIN] bf16 (D|N|F rows)
    const float* __restrict__ b_dnf, // [3S]
    float* __restrict__ st_f32,      // [B, S] in: st, out: st_new (in-place)
    u16* __restrict__ st_out_bf)     // [B, S] bf16 copy of st_new
{
    const int lane = threadIdx.x & 31;
    const int wid  = threadIdx.x >> 5;
    const int M0 = blockIdx.x * 64 + (wid >> 2) * 32;
    const int J0 = blockIdx.y * 64 + (wid & 3) * 16;

    v8f d0 = {}, d1 = {}, n0 = {}, n1 = {}, f0 = {}, f1 = {};
    for (int k0 = 0; k0 < CIN; k0 += 32) {
        const u16* abase; int astride, kk;
        if (k0 < SDIM) {
            abase = st_bf;                astride = SDIM;          kk = k0;
        } else if (k0 < SDIM + ADIM) {
            abase = act_bf + t * ADIM;    astride = DELAYN * ADIM; kk = k0 - SDIM;
        } else {
            abase = hh_bf;                astride = SDIM;          kk = k0 - SDIM - ADIM;
        }
        v16bf a0 = load_a(abase, astride, M0,      kk, lane);
        v16bf a1 = load_a(abase, astride, M0 + 16, kk, lane);
        v16bf bD = load_b(Wdnf, CIN, J0,            k0, lane);
        v16bf bN = load_b(Wdnf, CIN, J0 + SDIM,     k0, lane);
        v16bf bF = load_b(Wdnf, CIN, J0 + 2 * SDIM, k0, lane);
        d0 = wmma_bf16(a0, bD, d0);  d1 = wmma_bf16(a1, bD, d1);
        n0 = wmma_bf16(a0, bN, n0);  n1 = wmma_bf16(a1, bN, n1);
        f0 = wmma_bf16(a0, bF, f0);  f1 = wmma_bf16(a1, bF, f1);
    }

    const int nn = lane & 15;
    const float bD = b_dnf[J0 + nn];
    const float bN = b_dnf[SDIM + J0 + nn];
    const float bF = b_dnf[2 * SDIM + J0 + nn];
#pragma unroll
    for (int tile = 0; tile < 2; ++tile) {
        const v8f& aD = tile ? d1 : d0;
        const v8f& aN = tile ? n1 : n0;
        const v8f& aF = tile ? f1 : f0;
        const int mbase = M0 + tile * 16 + (lane >> 4) * 8;
#pragma unroll
        for (int v = 0; v < 8; ++v) {
            size_t idx = (size_t)(mbase + v) * SDIM + J0 + nn;
            float D = aD[v] + bD;
            float N = aN[v] + bN;
            float F = aF[v] + bF;
            float s = st_f32[idx];
            float sn = F * (s + D) + (1.f - F) * N;
            st_f32[idx]    = sn;
            st_out_bf[idx] = f2bf(sn);
        }
    }
}

// ---------------------------------------------------------------------------
// Launcher: prep (bf16 conversions, concat, init) + 33 GEMM kernels on stream.
// Kernel boundaries provide the grid-wide sync the recurrence needs.
// ---------------------------------------------------------------------------
extern "C" void kernel_launch(void* const* d_in, const int* in_sizes, int n_in,
                              void* d_out, int out_size, void* d_ws, size_t ws_size,
                              hipStream_t stream) {
    const float* state   = (const float*)d_in[0];   // [B, S]
    const float* action  = (const float*)d_in[1];   // [B, DELAY, A]
    const float* hidden0 = (const float*)d_in[2];   // [S]
    const float* W_ih    = (const float*)d_in[3];   // [3S, S]
    const float* W_hh    = (const float*)d_in[4];   // [3S, S]
    const float* b_ih    = (const float*)d_in[5];   // [3S]
    const float* b_hh    = (const float*)d_in[6];   // [3S]
    const float* W_D     = (const float*)d_in[7];   // [S, CIN]
    const float* b_D     = (const float*)d_in[8];   // [S]
    const float* W_N     = (const float*)d_in[9];
    const float* b_N     = (const float*)d_in[10];
    const float* W_F     = (const float*)d_in[11];
    const float* b_F     = (const float*)d_in[12];

    float* h_f32 = (float*)d_out;                   // rolling hidden, final answer

    // ---- workspace carve-out (256B aligned) ----
    char* ws = (char*)d_ws;
    size_t off = 0;
    auto take = [&](size_t bytes) -> char* {
        char* p = ws + off;
        off += (bytes + 255) & ~(size_t)255;
        return p;
    };
    u16*   Wih_bf  = (u16*)  take((size_t)G3 * SDIM * 2);
    u16*   Whh_bf  = (u16*)  take((size_t)G3 * SDIM * 2);
    u16*   Wdnf_bf = (u16*)  take((size_t)G3 * CIN * 2);
    float* bdnf    = (float*)take((size_t)G3 * 4);
    u16*   act_bf  = (u16*)  take((size_t)BATCH * DELAYN * ADIM * 2);
    float* st_f32  = (float*)take((size_t)BATCH * SDIM * 4);
    u16*   st_bfA  = (u16*)  take((size_t)BATCH * SDIM * 2);
    u16*   st_bfB  = (u16*)  take((size_t)BATCH * SDIM * 2);
    u16*   h_bfA   = (u16*)  take((size_t)BATCH * SDIM * 2);
    u16*   h_bfB   = (u16*)  take((size_t)BATCH * SDIM * 2);
    (void)ws_size; (void)n_in; (void)in_sizes; (void)out_size;

    const int THR = 256;
    auto nb = [](int n) { return (n + 255) / 256; };

    // ---- prep: weight/bias/activation conversion ----
    f32_to_bf16_kernel<<<nb(G3 * SDIM), THR, 0, stream>>>(W_ih, Wih_bf, G3 * SDIM);
    f32_to_bf16_kernel<<<nb(G3 * SDIM), THR, 0, stream>>>(W_hh, Whh_bf, G3 * SDIM);
    f32_to_bf16_kernel<<<nb(SDIM * CIN), THR, 0, stream>>>(W_D, Wdnf_bf,                          SDIM * CIN);
    f32_to_bf16_kernel<<<nb(SDIM * CIN), THR, 0, stream>>>(W_N, Wdnf_bf + (size_t)SDIM * CIN,     SDIM * CIN);
    f32_to_bf16_kernel<<<nb(SDIM * CIN), THR, 0, stream>>>(W_F, Wdnf_bf + (size_t)2 * SDIM * CIN, SDIM * CIN);
    copy_f32_kernel<<<nb(SDIM), THR, 0, stream>>>(b_D, bdnf,            SDIM);
    copy_f32_kernel<<<nb(SDIM), THR, 0, stream>>>(b_N, bdnf + SDIM,     SDIM);
    copy_f32_kernel<<<nb(SDIM), THR, 0, stream>>>(b_F, bdnf + 2 * SDIM, SDIM);
    f32_to_bf16_kernel<<<nb(BATCH * DELAYN * ADIM), THR, 0, stream>>>(action, act_bf, BATCH * DELAYN * ADIM);
    f32_to_bf16_kernel<<<nb(BATCH * SDIM), THR, 0, stream>>>(state, st_bfA, BATCH * SDIM);
    copy_f32_kernel<<<nb(BATCH * SDIM), THR, 0, stream>>>(state, st_f32, BATCH * SDIM);
    init_h_kernel<<<nb(BATCH * SDIM), THR, 0, stream>>>(hidden0, h_f32, h_bfA, BATCH * SDIM);

    // ---- recurrence ----
    dim3 blk(256);
    dim3 grd(BATCH / 64, SDIM / 64);   // (64, 8)

    // initial GRU: h1 = gru(state, h0)
    gru_kernel<<<grd, blk, 0, stream>>>(st_bfA, h_bfA, Wih_bf, Whh_bf,
                                        b_ih, b_hh, h_f32, h_bfB);
    u16* hb_cur = h_bfB;  u16* hb_nxt = h_bfA;
    u16* sb_cur = st_bfA; u16* sb_nxt = st_bfB;

    for (int t = 0; t < DELAYN; ++t) {
        dnf_kernel<<<grd, blk, 0, stream>>>(sb_cur, act_bf, t, hb_cur,
                                            Wdnf_bf, bdnf, st_f32, sb_nxt);
        { u16* tmp = sb_cur; sb_cur = sb_nxt; sb_nxt = tmp; }
        gru_kernel<<<grd, blk, 0, stream>>>(sb_cur, hb_cur, Wih_bf, Whh_bf,
                                            b_ih, b_hh, h_f32, hb_nxt);
        { u16* tmp = hb_cur; hb_cur = hb_nxt; hb_nxt = tmp; }
    }
    // h_f32 == d_out already holds the final hidden state.
}
